// STGAEGATTransformer_55353538511018
// MI455X (gfx1250) — compile-verified, matrix operand order checked by reading, not run on hardware
//
#include <hip/hip_runtime.h>
#include <hip/hip_bf16.h>
#include <math.h>

typedef __attribute__((ext_vector_type(16))) _Float16 v16h;
typedef __attribute__((ext_vector_type(8)))  _Float16 h8;
typedef __attribute__((ext_vector_type(4)))  _Float16 h4;
typedef __attribute__((ext_vector_type(2)))  _Float16 h2;
typedef __attribute__((ext_vector_type(8)))  float    v8f;
typedef __attribute__((ext_vector_type(4)))  float    f4;

#define Bdim 256
#define Tdim 72
#define Ndim 5
#define Fdim 10
#define Ddim 128
#define FFdim 256
#define NHdim 4
#define DHdim 32

// ---------------------------------------------------------------------------
// 1. Normalized adjacency (5x5) from edge_index + self loops
// ---------------------------------------------------------------------------
__global__ void k_anorm(const int* __restrict__ ei, int E, float* __restrict__ anorm) {
    if (threadIdx.x != 0 || blockIdx.x != 0) return;
    float A[25];
    for (int i = 0; i < 25; ++i) A[i] = 0.f;
    for (int e = 0; e < E; ++e) {
        int s = ei[e], d = ei[E + e];
        A[d * 5 + s] += 1.f;
    }
    for (int i = 0; i < 5; ++i) A[i * 5 + i] += 1.f;
    float dinv[5];
    for (int i = 0; i < 5; ++i) {
        float deg = 0.f;
        for (int j = 0; j < 5; ++j) deg += A[i * 5 + j];
        dinv[i] = (deg > 0.f) ? rsqrtf(deg) : 0.f;
    }
    for (int i = 0; i < 5; ++i)
        for (int j = 0; j < 5; ++j)
            anorm[i * 5 + j] = dinv[i] * A[i * 5 + j] * dinv[j];
}

// ---------------------------------------------------------------------------
// 2. GCN: tanh(Anorm @ (x@W + b))   one block per (b,t), 320 threads
// ---------------------------------------------------------------------------
__global__ void k_gcn(const float* __restrict__ x, const float* __restrict__ gcn_w,
                      const float* __restrict__ gcn_b, const float* __restrict__ anorm,
                      float* __restrict__ out) {
    int bt  = blockIdx.x;
    int tid = threadIdx.x;
    int n = tid >> 6, c = tid & 63;
    __shared__ float xw[Ndim][64];
    __shared__ float An[25];
    if (tid < 25) An[tid] = anorm[tid];
    const float* xp = x + ((size_t)bt * Ndim + n) * Fdim;
    float s = gcn_b[c];
#pragma unroll
    for (int f = 0; f < Fdim; ++f) s += xp[f] * gcn_w[f * 64 + c];
    xw[n][c] = s;
    __syncthreads();
    float acc = 0.f;
#pragma unroll
    for (int j = 0; j < Ndim; ++j) acc += An[n * 5 + j] * xw[j][c];
    out[((size_t)bt * Ndim + n) * 64 + c] = tanhf(acc);
}

// ---------------------------------------------------------------------------
// 3. LDS-staged WMMA GEMM: C[MxN] = A[MxK] @ B[KxN] (+bias, optional ReLU)
//    Block tile 64x128, BK=32, 256 threads = 8 waves (2M x 4N wave grid).
//    Each wave: 32x32 register tile = 2x2 wmma per K-step (A/B frags reused).
//    Staging: all 6 global_load_b128 issued as a clause, then convert+store.
//    Requires: M%64==0, N%128==0, K%32==0 (guaranteed by caller).
// ---------------------------------------------------------------------------
#define LDS_STRIDE 40   // halves; multiple of 8 -> 16B-aligned b128 chunks
__global__ void k_gemm_wmma(const float* __restrict__ A, const float* __restrict__ B,
                            const float* __restrict__ bias, float* __restrict__ C,
                            int M, int K, int N, int relu) {
    __shared__ __align__(16) _Float16 As[64][LDS_STRIDE];
    __shared__ __align__(16) _Float16 Bs[128][LDS_STRIDE];  // transposed: Bs[n][k]
    int tid  = threadIdx.x;
    int lane = tid & 31, wave = tid >> 5;
    int tilesN = N >> 7;
    int tm = (blockIdx.x / tilesN) << 6;
    int tn = (blockIdx.x % tilesN) << 7;
    int waveM = wave >> 2, waveN = wave & 3;                // 2 x 4
    int half16 = lane >> 4, idx16 = lane & 15;

    // ---- per-thread staging addresses, hoisted out of the K-loop ----
    // A chunk (64 x 32): 512 float4, thread covers f = tid and tid+256
    int fa0 = tid, fa1 = tid + 256;
    const float* apt0 = A + (size_t)(tm + (fa0 >> 3)) * K + (fa0 & 7) * 4;
    const float* apt1 = A + (size_t)(tm + (fa1 >> 3)) * K + (fa1 & 7) * 4;
    _Float16* ast0 = &As[fa0 >> 3][(fa0 & 7) * 4];
    _Float16* ast1 = &As[fa1 >> 3][(fa1 & 7) * 4];
    // B chunk (32 x 128): 512 (k-pair, n4) combos
    int fb0 = tid, fb1 = tid + 256;
    int kr0 = (fb0 >> 5) * 2, n40 = fb0 & 31;
    int kr1 = (fb1 >> 5) * 2, n41 = fb1 & 31;
    const float* bpt0 = B + (size_t)kr0 * N + tn + n40 * 4;
    const float* bpt1 = B + (size_t)kr1 * N + tn + n41 * 4;

    v8f acc00 = {}, acc01 = {}, acc10 = {}, acc11 = {};
    for (int kk = 0; kk < K; kk += 32) {
        // ---- issue all staging loads as one clause ----
        f4 va0  = *(const f4*)apt0;
        f4 va1  = *(const f4*)apt1;
        f4 vb00 = *(const f4*)bpt0;
        f4 vb01 = *(const f4*)(bpt0 + N);
        f4 vb10 = *(const f4*)bpt1;
        f4 vb11 = *(const f4*)(bpt1 + N);
        apt0 += 32; apt1 += 32;
        bpt0 += (size_t)32 * N; bpt1 += (size_t)32 * N;

        // ---- convert + LDS stores ----
        {
            h4 hv;
            hv[0] = (_Float16)va0[0]; hv[1] = (_Float16)va0[1];
            hv[2] = (_Float16)va0[2]; hv[3] = (_Float16)va0[3];
            *(h4*)ast0 = hv;
            hv[0] = (_Float16)va1[0]; hv[1] = (_Float16)va1[1];
            hv[2] = (_Float16)va1[2]; hv[3] = (_Float16)va1[3];
            *(h4*)ast1 = hv;
        }
#pragma unroll
        for (int j = 0; j < 4; ++j) {
            h2 p;
            p[0] = (_Float16)vb00[j]; p[1] = (_Float16)vb01[j];
            *(h2*)&Bs[n40 * 4 + j][kr0] = p;
            p[0] = (_Float16)vb10[j]; p[1] = (_Float16)vb11[j];
            *(h2*)&Bs[n41 * 4 + j][kr1] = p;
        }
        __syncthreads();

        // prefetch next K-chunk into cache (gfx1250 global_prefetch path)
        if (kk + 32 < K) {
            __builtin_prefetch(apt0, 0, 1);
            __builtin_prefetch(bpt0, 0, 1);
        }

        // ---- fragments from LDS: two 16B ds_load_b128 per operand ----
        // ISA f16 A layout: lanes 0-15 hold K {0..7,16..23}, lanes 16-31 {8..15,24..31}
        v16h af0, af1, bf0, bf1;
        {
            int r0 = waveM * 32 + idx16;
            h8 lo0 = *(const h8*)&As[r0][half16 * 8];
            h8 hi0 = *(const h8*)&As[r0][16 + half16 * 8];
            h8 lo1 = *(const h8*)&As[r0 + 16][half16 * 8];
            h8 hi1 = *(const h8*)&As[r0 + 16][16 + half16 * 8];
            int n0 = waveN * 32 + idx16;
            h8 blo0 = *(const h8*)&Bs[n0][half16 * 8];
            h8 bhi0 = *(const h8*)&Bs[n0][16 + half16 * 8];
            h8 blo1 = *(const h8*)&Bs[n0 + 16][half16 * 8];
            h8 bhi1 = *(const h8*)&Bs[n0 + 16][16 + half16 * 8];
#pragma unroll
            for (int h = 0; h < 8; ++h) {
                af0[h] = lo0[h];  af0[h + 8] = hi0[h];
                af1[h] = lo1[h];  af1[h + 8] = hi1[h];
                bf0[h] = blo0[h]; bf0[h + 8] = bhi0[h];
                bf1[h] = blo1[h]; bf1[h + 8] = bhi1[h];
            }
        }
        acc00 = __builtin_amdgcn_wmma_f32_16x16x32_f16(false, af0, false, bf0,
                                                       (short)0, acc00, false, false);
        acc01 = __builtin_amdgcn_wmma_f32_16x16x32_f16(false, af0, false, bf1,
                                                       (short)0, acc01, false, false);
        acc10 = __builtin_amdgcn_wmma_f32_16x16x32_f16(false, af1, false, bf0,
                                                       (short)0, acc10, false, false);
        acc11 = __builtin_amdgcn_wmma_f32_16x16x32_f16(false, af1, false, bf1,
                                                       (short)0, acc11, false, false);
        __syncthreads();
    }

    // ---- epilogue: D layout n = lane%16, m = r + 8*(lane>=16) ----
    int col0 = tn + waveN * 32 + idx16;
    int col1 = col0 + 16;
    float bv0 = bias ? bias[col0] : 0.f;
    float bv1 = bias ? bias[col1] : 0.f;
    int mbase = tm + waveM * 32 + (half16 << 3);
#pragma unroll
    for (int r = 0; r < 8; ++r) {
        float v00 = acc00[r] + bv0, v01 = acc01[r] + bv1;
        float v10 = acc10[r] + bv0, v11 = acc11[r] + bv1;
        if (relu) {
            v00 = v00 > 0.f ? v00 : 0.f; v01 = v01 > 0.f ? v01 : 0.f;
            v10 = v10 > 0.f ? v10 : 0.f; v11 = v11 > 0.f ? v11 : 0.f;
        }
        C[(size_t)(mbase + r) * N + col0]      = v00;
        C[(size_t)(mbase + r) * N + col1]      = v01;
        C[(size_t)(mbase + r + 16) * N + col0] = v10;
        C[(size_t)(mbase + r + 16) * N + col1] = v11;
    }
}

// ---------------------------------------------------------------------------
// 4. GAT attention + aggregation + LayerNorm + node-mean + positional encoding
// ---------------------------------------------------------------------------
__global__ void k_gat(const float* __restrict__ hbuf, const int* __restrict__ ei, int E,
                      const float* __restrict__ a_src, const float* __restrict__ a_dst,
                      const float* __restrict__ gat_b,
                      const float* __restrict__ gln_g, const float* __restrict__ gln_b,
                      float* __restrict__ zout) {
    int bt  = blockIdx.x;
    int t_idx = bt % Tdim;
    int tid = threadIdx.x;
    int EE = E + Ndim;
    __shared__ float hs[Ndim][Ddim];
    __shared__ int   s2[32], d2[32];
    __shared__ float as_[Ndim][NHdim], ad_[Ndim][NHdim];
    __shared__ float ee[32][NHdim];
    __shared__ float mx[Ndim][NHdim], den[Ndim][NHdim];
    __shared__ float agg[Ndim][Ddim];
    __shared__ float mu[Ndim], rs[Ndim];

    if (tid < E)              { s2[tid] = ei[tid]; d2[tid] = ei[E + tid]; }
    if (tid >= E && tid < EE) { s2[tid] = tid - E; d2[tid] = tid - E;     }
    for (int i = tid; i < Ndim * Ddim; i += 128)
        hs[i >> 7][i & 127] = hbuf[(size_t)bt * (Ndim * Ddim) + i];
    __syncthreads();

    if (tid < Ndim * NHdim) {
        int n = tid >> 2, hh = tid & 3;
        float s = 0.f, d = 0.f;
        for (int c = 0; c < DHdim; ++c) {
            float v = hs[n][hh * DHdim + c];
            s += v * a_src[hh * DHdim + c];
            d += v * a_dst[hh * DHdim + c];
        }
        as_[n][hh] = s; ad_[n][hh] = d;
    }
    __syncthreads();

    if (tid < EE * NHdim) {
        int e = tid >> 2, hh = tid & 3;
        float v = as_[s2[e]][hh] + ad_[d2[e]][hh];
        ee[e][hh] = v > 0.f ? v : 0.2f * v;
    }
    __syncthreads();

    if (tid < Ndim * NHdim) {
        int n = tid >> 2, hh = tid & 3;
        float m = -1e30f;
        for (int e = 0; e < EE; ++e) if (d2[e] == n) m = fmaxf(m, ee[e][hh]);
        float dv = 0.f;
        for (int e = 0; e < EE; ++e) if (d2[e] == n) dv += expf(ee[e][hh] - m);
        mx[n][hh] = m; den[n][hh] = dv;
    }
    __syncthreads();

    for (int i = tid; i < Ndim * Ddim; i += 128) {
        int n = i >> 7, ch = i & 127, hh = ch >> 5;
        float acc = 0.f;
        for (int e = 0; e < EE; ++e) {
            if (d2[e] == n) {
                float alpha = expf(ee[e][hh] - mx[n][hh]) / den[n][hh];
                acc += alpha * hs[s2[e]][ch];
            }
        }
        agg[n][ch] = acc + gat_b[ch];
    }
    __syncthreads();

    if (tid < Ndim) {
        float m = 0.f;
        for (int ch = 0; ch < Ddim; ++ch) m += agg[tid][ch];
        m /= (float)Ddim;
        float v = 0.f;
        for (int ch = 0; ch < Ddim; ++ch) { float d = agg[tid][ch] - m; v += d * d; }
        v /= (float)Ddim;
        mu[tid] = m; rs[tid] = rsqrtf(v + 1e-5f);
    }
    __syncthreads();

    if (tid < Ddim) {
        float zs = 0.f;
        for (int n = 0; n < Ndim; ++n)
            zs += (agg[n][tid] - mu[n]) * rs[n] * gln_g[tid] + gln_b[tid];
        zs *= (1.f / (float)Ndim);
        int i2 = tid >> 1;
        float div = expf((float)(2 * i2) * (-9.210340371976184f / (float)Ddim));
        float ang = (float)t_idx * div;
        zs += (tid & 1) ? cosf(ang) : sinf(ang);
        zout[(size_t)bt * Ddim + tid] = zs;
    }
}

// ---------------------------------------------------------------------------
// 5. Attention: one block per (b, head); K/V in LDS, online softmax per row
// ---------------------------------------------------------------------------
__global__ void k_attn(const float* __restrict__ qkv, float* __restrict__ attout) {
    int b  = blockIdx.x >> 2;
    int hh = blockIdx.x & 3;
    int tid = threadIdx.x;
    __shared__ float Ks[Tdim][DHdim + 1];
    __shared__ float Vs[Tdim][DHdim + 1];
    const float* base = qkv + (size_t)b * Tdim * (3 * Ddim);
    for (int i = tid; i < Tdim * DHdim; i += blockDim.x) {
        int r = i >> 5, c = i & 31;
        Ks[r][c] = base[(size_t)r * (3 * Ddim) + Ddim     + hh * DHdim + c];
        Vs[r][c] = base[(size_t)r * (3 * Ddim) + 2 * Ddim + hh * DHdim + c];
    }
    __syncthreads();
    if (tid < Tdim) {
        float q[DHdim], o[DHdim];
#pragma unroll
        for (int c = 0; c < DHdim; ++c) {
            q[c] = base[(size_t)tid * (3 * Ddim) + hh * DHdim + c];
            o[c] = 0.f;
        }
        float m = -1e30f, l = 0.f;
        const float inv_scale = 0.17677669529663687f;   // 1/sqrt(32)
        for (int j = 0; j < Tdim; ++j) {
            float s = 0.f;
#pragma unroll
            for (int c = 0; c < DHdim; ++c) s += q[c] * Ks[j][c];
            s *= inv_scale;
            float mn   = fmaxf(m, s);
            float corr = expf(m - mn);
            float p    = expf(s - mn);
            l = l * corr + p;
#pragma unroll
            for (int c = 0; c < DHdim; ++c) o[c] = o[c] * corr + p * Vs[j][c];
            m = mn;
        }
        float inv_l = 1.f / l;
        float* op = attout + ((size_t)b * Tdim + tid) * Ddim + hh * DHdim;
#pragma unroll
        for (int c = 0; c < DHdim; ++c) op[c] = o[c] * inv_l;
    }
}

// ---------------------------------------------------------------------------
// 6. Residual add + LayerNorm over 128, one wave32 per row
// ---------------------------------------------------------------------------
__global__ void k_ln_res(float* __restrict__ z, const float* __restrict__ add,
                         const float* __restrict__ g, const float* __restrict__ bt, int R) {
    int wave = threadIdx.x >> 5, lane = threadIdx.x & 31;
    int row = blockIdx.x * (blockDim.x >> 5) + wave;
    if (row >= R) return;
    float v[4];
    float s = 0.f;
#pragma unroll
    for (int k = 0; k < 4; ++k) {
        int ch = lane * 4 + k;
        v[k] = z[(size_t)row * Ddim + ch] + add[(size_t)row * Ddim + ch];
        s += v[k];
    }
#pragma unroll
    for (int off = 16; off > 0; off >>= 1) s += __shfl_xor(s, off, 32);
    float mu = s / (float)Ddim;
    float var = 0.f;
#pragma unroll
    for (int k = 0; k < 4; ++k) { float d = v[k] - mu; var += d * d; }
#pragma unroll
    for (int off = 16; off > 0; off >>= 1) var += __shfl_xor(var, off, 32);
    float rsg = rsqrtf(var / (float)Ddim + 1e-5f);
#pragma unroll
    for (int k = 0; k < 4; ++k) {
        int ch = lane * 4 + k;
        z[(size_t)row * Ddim + ch] = (v[k] - mu) * rsg * g[ch] + bt[ch];
    }
}

// ---------------------------------------------------------------------------
// 7. Output heads: 6 small MLPs on z[:, T-1]
// ---------------------------------------------------------------------------
__global__ void k_head(const float* __restrict__ z,
                       const float* __restrict__ hw1, const float* __restrict__ hb1,
                       const float* __restrict__ hw2, const float* __restrict__ hb2,
                       float* __restrict__ out) {
    int k = blockIdx.x / Bdim;
    int b = blockIdx.x % Bdim;
    int tid = threadIdx.x;                   // 64 threads
    __shared__ float h1[64];
    const float* last = z + ((size_t)b * Tdim + (Tdim - 1)) * Ddim;
    float s = hb1[k * 64 + tid];
    for (int d = 0; d < Ddim; ++d) s += last[d] * hw1[((size_t)k * Ddim + d) * 64 + tid];
    h1[tid] = s > 0.f ? s : 0.f;
    __syncthreads();
    if (tid < 5) {
        float o = hb2[k * 5 + tid];
        for (int m = 0; m < 64; ++m) o += h1[m] * hw2[((size_t)k * 64 + m) * 5 + tid];
        out[((size_t)k * Bdim + b) * 5 + tid] = o;
    }
}

// ---------------------------------------------------------------------------
// Launcher
// ---------------------------------------------------------------------------
static inline void launch_gemm(const float* A, const float* B, const float* bias,
                               float* C, int M, int K, int N, int relu,
                               hipStream_t stream) {
    int blocks = (M / 64) * (N / 128);
    hipLaunchKernelGGL(k_gemm_wmma, dim3(blocks), dim3(256), 0, stream,
                       A, B, bias, C, M, K, N, relu);
}

extern "C" void kernel_launch(void* const* d_in, const int* in_sizes, int n_in,
                              void* d_out, int out_size, void* d_ws, size_t ws_size,
                              hipStream_t stream) {
    (void)n_in; (void)out_size; (void)ws_size;
    const float* x        = (const float*)d_in[0];
    const int*   ei       = (const int*)  d_in[1];
    const float* gcn_w    = (const float*)d_in[2];
    const float* gcn_b    = (const float*)d_in[3];
    const float* gat_w    = (const float*)d_in[4];
    const float* gat_asrc = (const float*)d_in[5];
    const float* gat_adst = (const float*)d_in[6];
    const float* gat_b    = (const float*)d_in[7];
    const float* gln_g    = (const float*)d_in[8];
    const float* gln_b    = (const float*)d_in[9];
    const float* tw_qkv   = (const float*)d_in[10];
    const float* tb_qkv   = (const float*)d_in[11];
    const float* tw_o     = (const float*)d_in[12];
    const float* tb_o     = (const float*)d_in[13];
    const float* ln1_g    = (const float*)d_in[14];
    const float* ln1_b    = (const float*)d_in[15];
    const float* w_ff1    = (const float*)d_in[16];
    const float* b_ff1    = (const float*)d_in[17];
    const float* w_ff2    = (const float*)d_in[18];
    const float* b_ff2    = (const float*)d_in[19];
    const float* ln2_g    = (const float*)d_in[20];
    const float* ln2_b    = (const float*)d_in[21];
    const float* hw1      = (const float*)d_in[22];
    const float* hb1      = (const float*)d_in[23];
    const float* hw2      = (const float*)d_in[24];
    const float* hb2      = (const float*)d_in[25];
    float* out = (float*)d_out;

    const int E  = in_sizes[1] / 2;           // 20 directed edges
    const size_t BT = (size_t)Bdim * Tdim;    // 18432

    float* ws      = (float*)d_ws;
    float* anorm   = ws;                      // 25 floats (padded to 64)
    float* buf     = ws + 64;
    // phase 1 (graph):
    float* buf_gcn = buf;                      // BT*5*64
    float* buf_h   = buf_gcn + BT * Ndim * 64; // BT*5*128
    float* buf_z   = buf_h + BT * Ndim * 128;  // BT*128  (persists)
    // phase 2 (transformer) aliases phase-1 scratch:
    float* buf_qkv = buf;                      // BT*384
    float* buf_att = buf_qkv + BT * 384;       // BT*128
    float* buf_tmp = buf_att + BT * 128;       // BT*128
    float* buf_ff1 = buf_tmp + BT * 128;       // BT*256 (ends below buf_z)

    // --- graph phase ---
    hipLaunchKernelGGL(k_anorm, dim3(1), dim3(32), 0, stream, ei, E, anorm);
    hipLaunchKernelGGL(k_gcn, dim3((unsigned)BT), dim3(Ndim * 64), 0, stream,
                       x, gcn_w, gcn_b, anorm, buf_gcn);
    // H = gcn @ gat_w : M=92160, K=64, N=128
    launch_gemm(buf_gcn, gat_w, nullptr, buf_h, (int)BT * Ndim, 64, 128, 0, stream);
    hipLaunchKernelGGL(k_gat, dim3((unsigned)BT), dim3(128), 0, stream,
                       buf_h, ei, E, gat_asrc, gat_adst, gat_b, gln_g, gln_b, buf_z);

    // --- transformer phase ---
    const int M = (int)BT;                    // 18432
    for (int i = 0; i < 3; ++i) {
        launch_gemm(buf_z, tw_qkv + (size_t)i * Ddim * 3 * Ddim,
                    tb_qkv + (size_t)i * 3 * Ddim, buf_qkv, M, Ddim, 3 * Ddim, 0, stream);
        hipLaunchKernelGGL(k_attn, dim3(Bdim * NHdim), dim3(96), 0, stream, buf_qkv, buf_att);
        launch_gemm(buf_att, tw_o + (size_t)i * Ddim * Ddim,
                    tb_o + (size_t)i * Ddim, buf_tmp, M, Ddim, Ddim, 0, stream);
        hipLaunchKernelGGL(k_ln_res, dim3((M + 3) / 4), dim3(128), 0, stream,
                           buf_z, buf_tmp, ln1_g + (size_t)i * Ddim, ln1_b + (size_t)i * Ddim, M);
        launch_gemm(buf_z, w_ff1 + (size_t)i * Ddim * FFdim,
                    b_ff1 + (size_t)i * FFdim, buf_ff1, M, Ddim, FFdim, 1, stream);
        launch_gemm(buf_ff1, w_ff2 + (size_t)i * FFdim * Ddim,
                    b_ff2 + (size_t)i * Ddim, buf_tmp, M, FFdim, Ddim, 0, stream);
        hipLaunchKernelGGL(k_ln_res, dim3((M + 3) / 4), dim3(128), 0, stream,
                           buf_z, buf_tmp, ln2_g + (size_t)i * Ddim, ln2_b + (size_t)i * Ddim, M);
    }

    // --- heads ---
    hipLaunchKernelGGL(k_head, dim3(6 * Bdim), dim3(64), 0, stream,
                       buf_z, hw1, hb1, hw2, hb2, out);
}
